// FDConvBlock_52621939311408
// MI455X (gfx1250) — matrix operand order repaired and checked by reading, not compiled, not run on hardware
//
#include <hip/hip_runtime.h>

#define HH 320
#define WW 320
#define NB 8
#define COUT 64
#define HWPIX (HH*WW)   // 102400

typedef __attribute__((ext_vector_type(16))) __bf16 v16bf;
typedef __attribute__((ext_vector_type(8)))  float  v8f;
typedef int vint4 __attribute__((__vector_size__(16)));

union FragB16 { v16bf v; uint4 q[2]; };

#if defined(__has_builtin)
#if __has_builtin(__builtin_amdgcn_global_load_async_to_lds_b128) && \
    __has_builtin(__builtin_amdgcn_s_wait_asynccnt)
#define USE_ASYNC_LDS 1
#endif
#endif

__device__ __forceinline__ unsigned short f32_to_bf16(float f) {
  union { float f; unsigned int u; } v; v.f = f;
  unsigned int u = v.u;
  u += 0x7FFFu + ((u >> 16) & 1u);       // round-to-nearest-even
  return (unsigned short)(u >> 16);
}
__device__ __forceinline__ float bf16_to_f32(unsigned short u) {
  union { unsigned int i; float f; } v; v.i = ((unsigned int)u) << 16; return v.f;
}

// ---------------- weight pack: [O][I][3][3] f32 -> [O][ (ky*3+kx)*Cin + c ] bf16
__global__ void pack_w_kernel(const float* __restrict__ w, unsigned short* __restrict__ wp, int Cin) {
  int total = COUT * Cin * 9;
  int i = blockIdx.x * 256 + threadIdx.x;
  if (i >= total) return;
  int n = i / (Cin * 9);
  int r = i % (Cin * 9);
  int t = r / Cin;            // ky*3+kx
  int c = r % Cin;
  wp[n * (Cin * 9) + t * Cin + c] = f32_to_bf16(w[(n * Cin + c) * 9 + t]);
}

// ---------------- x: NCHW f32 -> NHWC bf16 (LDS transpose, coalesced both sides)
__global__ __launch_bounds__(256) void cvt_x_kernel(const float* __restrict__ x,
                                                    unsigned short* __restrict__ xp) {
  __shared__ float t[32 * 65];
  const int tid = threadIdx.x;
  const int bx = blockIdx.x;          // 8*320*5 = 12800 blocks
  const int w0 = (bx % 5) * 64;
  const int h  = (bx / 5) % HH;
  const int b  = bx / 1600;
  #pragma unroll
  for (int k = 0; k < 8; ++k) {
    int idx = tid + k * 256;          // 0..2047
    int c = idx >> 6, i = idx & 63;
    t[c * 65 + i] = x[(((size_t)b * 32 + c) * HH + h) * WW + w0 + i];
  }
  __syncthreads();
  #pragma unroll
  for (int k = 0; k < 8; ++k) {
    int idx = tid + k * 256;
    int c = idx & 31, i = idx >> 5;
    xp[((((size_t)b * HH + h) * WW + w0 + i) * 32) + c] = f32_to_bf16(t[c * 65 + i]);
  }
}

// ---------------- implicit-GEMM 3x3 conv, bf16 WMMA, NHWC in/out
// Block tile: 64 pixels (contiguous w, one h row) x 64 Cout. 8 waves, each 16M x 32N.
// Halo input tile (3 rows x 66 pixels x CIN) is staged to LDS ONCE (async if
// available), then the whole K loop (9 or 18 steps) runs with no barriers.
template <int CIN>
__global__ __launch_bounds__(256) void conv_gemm_kernel(
    const unsigned short* __restrict__ in,   // NHWC bf16 [B][H][W][CIN]
    const unsigned short* __restrict__ wp,   // [COUT][CIN*9] bf16, K-major
    unsigned short* __restrict__ out)        // NHWC bf16 [B][H][W][COUT]
{
  constexpr int KTOT = CIN * 9;
  constexpr int PSTR = CIN + 8;              // padded pixel stride (halfwords) -> conflict-free ds_load_b128
  __shared__ unsigned short At[3 * 66 * PSTR];

  const int tid  = threadIdx.x;
  const int lane = tid & 31;
  const int wid  = tid >> 5;
  const int m0   = (wid & 3) * 16;
  const int n0   = (wid >> 2) * 32;

  const int bx = blockIdx.x;                 // 8*320*5 = 12800 blocks
  const int w0 = (bx % 5) * 64;
  const int h  = (bx / 5) % HH;
  const int b  = bx / 1600;

  // ---- one-shot halo staging: 3*66*CIN bf16, in 16B (8-channel) chunks
  constexpr int NCH = (3 * 66 * CIN) / 8;
  for (int k = tid; k < NCH; k += 256) {
    const int c8 = k % (CIN / 8);
    const int p  = (k / (CIN / 8)) % 66;
    const int r  = k / (66 * (CIN / 8));
    const int hh = h + r - 1;
    const int ww = w0 + p - 1;
    unsigned short* dst = &At[(r * 66 + p) * PSTR + c8 * 8];
    const bool valid = ((unsigned)hh < (unsigned)HH) && ((unsigned)ww < (unsigned)WW);
#if USE_ASYNC_LDS
    if (valid) {
      const unsigned short* src = in + ((((size_t)b * HH + hh) * WW + ww) * CIN) + c8 * 8;
      __builtin_amdgcn_global_load_async_to_lds_b128(
          (__attribute__((address_space(1))) vint4*)src,
          (__attribute__((address_space(3))) vint4*)dst, 0, 0);
    } else {
      *(uint4*)dst = make_uint4(0u, 0u, 0u, 0u);
    }
#else
    uint4 v = make_uint4(0u, 0u, 0u, 0u);
    if (valid) {
      const unsigned short* src = in + ((((size_t)b * HH + hh) * WW + ww) * CIN) + c8 * 8;
      v = *(const uint4*)src;
    }
    *(uint4*)dst = v;
#endif
  }
#if USE_ASYNC_LDS
  __builtin_amdgcn_s_wait_asynccnt(0);
#endif
  __syncthreads();                           // the only barrier in this kernel

  v8f acc[2];
  #pragma unroll
  for (int s = 0; s < 2; ++s)
    acc[s] = (v8f){0.f, 0.f, 0.f, 0.f, 0.f, 0.f, 0.f, 0.f};

  // ---- barrier-free WMMA K loop
  #pragma unroll
  for (int ky = 0; ky < 3; ++ky) {
    #pragma unroll
    for (int kx = 0; kx < 3; ++kx) {
      #pragma unroll
      for (int cb = 0; cb < CIN / 32; ++cb) {
        // A fragment: per-lane K runs 0-7 & 16-23 (low half) / 8-15 & 24-31 (high half)
        FragB16 a;
        const unsigned short* arow =
            &At[(ky * 66 + kx + m0 + (lane & 15)) * PSTR + cb * 32 + ((lane >> 4) * 8)];
        a.q[0] = *(const uint4*)(arow);
        a.q[1] = *(const uint4*)(arow + 16);

        const int ks = (ky * 3 + kx) * (CIN / 32) + cb;
        #pragma unroll
        for (int s = 0; s < 2; ++s) {
          FragB16 bf;
          const unsigned short* brow =
              wp + (size_t)(n0 + s * 16 + (lane & 15)) * KTOT + ks * 32 + ((lane >> 4) * 8);
          bf.q[0] = *(const uint4*)(brow);
          bf.q[1] = *(const uint4*)(brow + 16);
          acc[s] = __builtin_amdgcn_wmma_f32_16x16x32_bf16(
              false, a.v, false, bf.v, (short)0, acc[s], false, false);
        }
      }
    }
  }

  // Epilogue: C layout — lane<16: (M=r, N=lane); lane>=16: (M=r+8, N=lane-16)
  const int mb = w0 + m0 + ((lane >> 4) * 8);
  const size_t pixbase = ((size_t)b * HH + h) * WW;
  #pragma unroll
  for (int s = 0; s < 2; ++s) {
    const int n = n0 + s * 16 + (lane & 15);
    #pragma unroll
    for (int r = 0; r < 8; ++r) {
      out[(pixbase + mb + r) * COUT + n] = f32_to_bf16(acc[s][r]);
    }
  }
}

// ---------------- per-(b,c) sum / sumsq partials (coalesced), atomic merge
__global__ __launch_bounds__(256) void stats_part_kernel(const unsigned short* __restrict__ raw,
                                                         float* __restrict__ st) {
  const int tid = threadIdx.x;
  const int b = blockIdx.x / 100, chunk = blockIdx.x % 100;  // 100 chunks of 1024 pixels
  const int c = tid & 63, p0 = tid >> 6;                     // 4 pixels x 64 channels
  float s = 0.f, s2 = 0.f;
  const size_t base = (size_t)b * HWPIX + (size_t)chunk * 1024;
  for (int p = p0; p < 1024; p += 4) {
    float v = bf16_to_f32(raw[(base + p) * 64 + c]);
    s += v; s2 += v * v;
  }
  __shared__ float red[512];
  red[tid] = s; red[256 + tid] = s2;
  __syncthreads();
  if (tid < 64) {
    float ts  = red[tid] + red[tid + 64] + red[tid + 128] + red[tid + 192];
    float ts2 = red[256 + tid] + red[320 + tid] + red[384 + tid] + red[448 + tid];
    atomicAdd(&st[(b * 64 + tid) * 2 + 0], ts);
    atomicAdd(&st[(b * 64 + tid) * 2 + 1], ts2);
  }
}

__global__ void stats_fin_kernel(float* __restrict__ st) {
  int i = threadIdx.x;                         // 512 = 8*64 (b,c) pairs
  float s = st[i * 2], s2 = st[i * 2 + 1];
  float mean = s * (1.f / (float)HWPIX);
  float var  = s2 * (1.f / (float)HWPIX) - mean * mean;
  st[i * 2]     = mean;
  st[i * 2 + 1] = rsqrtf(var + 1e-5f);
}

// ---------------- normalize + ReLU in place (bf16 NHWC), vectorized 8 elements/thread
__global__ __launch_bounds__(256) void norm_relu_bf16_kernel(unsigned short* __restrict__ hbuf,
                                                             const float* __restrict__ st) {
  const size_t e0 = ((size_t)blockIdx.x * 256 + threadIdx.x) * 8;
  const int b  = (int)(e0 / ((size_t)HWPIX * 64));
  const int c0 = (int)(e0 & 63);
  uint4 v = *(uint4*)(hbuf + e0);
  unsigned short* u = (unsigned short*)&v;
  #pragma unroll
  for (int j = 0; j < 8; ++j) {
    int c = c0 + j;
    float mean = st[(b * 64 + c) * 2], inv = st[(b * 64 + c) * 2 + 1];
    float f = (bf16_to_f32(u[j]) - mean) * inv;
    u[j] = f32_to_bf16(fmaxf(f, 0.f));
  }
  *(uint4*)(hbuf + e0) = v;
}

// ---------------- normalize + ReLU + NHWC bf16 -> NCHW f32 (LDS transpose)
__global__ __launch_bounds__(256) void norm_relu_out_kernel(const unsigned short* __restrict__ raw,
                                                            const float* __restrict__ st,
                                                            float* __restrict__ out) {
  __shared__ float t[64 * 65];
  const int tid = threadIdx.x;
  const int bx = blockIdx.x;                 // 12800 blocks
  const int w0 = (bx % 5) * 64;
  const int h  = (bx / 5) % HH;
  const int b  = bx / 1600;
  const size_t base = (((size_t)b * HH + h) * WW + w0) * 64;
  #pragma unroll
  for (int k = 0; k < 16; ++k) {
    int idx = tid + k * 256;                 // 0..4095
    int i = idx >> 6, c = idx & 63;
    float mean = st[(b * 64 + c) * 2], inv = st[(b * 64 + c) * 2 + 1];
    float v = (bf16_to_f32(raw[base + (size_t)i * 64 + c]) - mean) * inv;
    t[c * 65 + i] = fmaxf(v, 0.f);
  }
  __syncthreads();
  #pragma unroll
  for (int k = 0; k < 16; ++k) {
    int idx = tid + k * 256;
    int i = idx & 63, c = idx >> 6;
    out[(((size_t)b * 64 + c) * HH + h) * WW + w0 + i] = t[c * 65 + i];
  }
}

extern "C" void kernel_launch(void* const* d_in, const int* in_sizes, int n_in,
                              void* d_out, int out_size, void* d_ws, size_t ws_size,
                              hipStream_t stream) {
  const float* x  = (const float*)d_in[0];
  const float* w1 = (const float*)d_in[1];
  const float* w2 = (const float*)d_in[3];   // d_in[2], d_in[4] are zero biases (no-op under InstanceNorm)
  float* out = (float*)d_out;

  char* ws = (char*)d_ws;
  // region0: x NHWC bf16 (52.4MB) early, reused as conv2 raw NHWC bf16 (104.9MB) later
  unsigned short* reg0 = (unsigned short*)(ws);
  unsigned short* hbuf = (unsigned short*)(ws + 104857600ull);              // 104.9MB
  unsigned short* w1p  = (unsigned short*)(ws + 209715200ull);              // 36KB
  unsigned short* w2p  = (unsigned short*)(ws + 209715200ull + 36864ull);   // 72KB
  float* st1 = (float*)(ws + 209715200ull + 36864ull + 73728ull);           // 4KB
  float* st2 = st1 + 1024;                                                  // 4KB

  (void)hipMemsetAsync(st1, 0, 2048 * sizeof(float), stream);

  pack_w_kernel<<<72, 256, 0, stream>>>(w1, w1p, 32);    // 64*32*9 = 18432
  pack_w_kernel<<<144, 256, 0, stream>>>(w2, w2p, 64);   // 64*64*9 = 36864
  cvt_x_kernel<<<12800, 256, 0, stream>>>(x, reg0);

  conv_gemm_kernel<32><<<12800, 256, 0, stream>>>(reg0, w1p, hbuf);
  stats_part_kernel<<<800, 256, 0, stream>>>(hbuf, st1);
  stats_fin_kernel<<<1, 512, 0, stream>>>(st1);
  norm_relu_bf16_kernel<<<25600, 256, 0, stream>>>(hbuf, st1);

  conv_gemm_kernel<64><<<12800, 256, 0, stream>>>(hbuf, w2p, reg0);
  stats_part_kernel<<<800, 256, 0, stream>>>(reg0, st2);
  stats_fin_kernel<<<1, 512, 0, stream>>>(st2);
  norm_relu_out_kernel<<<12800, 256, 0, stream>>>(reg0, st2, out);
}
